// TTTSwiGLU_89223650607440
// MI455X (gfx1250) — matrix operand order repaired and checked by reading, not compiled, not run on hardware
//
#include <hip/hip_runtime.h>
#include <hip/hip_bf16.h>

// ---------------------------------------------------------------------------
// TTT-SwiGLU for MI455X (gfx1250, wave32, WMMA bf16 -> f32 accum)
//
//   1. k_conv           : causal depthwise conv on ttt_target  -> t_bf (bf16)
//   2. k_gemm_swiglu    : h = silu(x Wg^T) * (x Wu^T)          -> h_bf (bf16)
//   3. k_gemm_tproj     : t_proj = t Wp^T (first 15 chunks)    -> tp_bf
//                         (A tile staged with GLOBAL_LOAD_ASYNC_TO_LDS_B128)
//   4. k_initW          : W[b] = w_down  (fp32 running weight)
//   5. per chunk s = 0..15 (stream-sequenced cumsum):
//        k_gemm_out(s)   : out_chunk = h_chunk @ W[b]^T  (A tile via TDM
//                          tensor_load_to_lds, one DMA per WG per K-step)
//        k_gemm_ddown(s) : d_tmp[b]  = t_proj_s^T @ h_s
//        k_axpy          : W += TTT_LR * d_tmp
// ---------------------------------------------------------------------------

typedef __bf16 bf16;
typedef __attribute__((ext_vector_type(16))) __bf16 v16bf;
typedef __attribute__((ext_vector_type(8)))  __bf16 v8bf;
typedef __attribute__((ext_vector_type(8)))  float  v8f;
typedef __attribute__((ext_vector_type(4)))  float  v4f;
typedef __attribute__((ext_vector_type(4)))  unsigned int v4u;
typedef __attribute__((ext_vector_type(8)))  int    v8i_;
typedef __attribute__((ext_vector_type(4)))  int    v4i_;

#define D_MODEL 1024
#define FFN     4096
#define CHUNK   256
#define NC      16
#define BATCH   2
#define SEQ     4096
#define TTT_LR  0.01f

// GEMM tiling: 128x64 block, BK=32, 8 waves (4 along M x 2 along N), 32x32/wave
#define BM 128
#define BN 64
#define BK 32
#define LDK 40          // BK + 8 bf16 pad -> 80B row stride, keeps 16B alignment
#define NTHREADS 256

// ---- CDNA5 async-copy helpers ---------------------------------------------
// Per-lane async copy 16B global -> LDS (tracked by ASYNCcnt).
// LDS operand is the wave-relative byte offset (low 32 bits of the flat addr).
__device__ __forceinline__ unsigned lds_off_of(const void* p) {
  return (unsigned)(reinterpret_cast<uintptr_t>(p));
}
__device__ __forceinline__ void async_copy_b128(const void* g, unsigned lds_off) {
  asm volatile("global_load_async_to_lds_b128 %0, %1, off"
               :: "v"(lds_off), "v"(g) : "memory");
}
__device__ __forceinline__ void wait_async0() {
  asm volatile("s_wait_asynccnt 0" ::: "memory");
}

// ---- WMMA fragment loaders (LDS -> VGPR, CDNA5 16-bit layouts) -------------
__device__ __forceinline__ v16bf frag_a(const bf16* As, int m0, int lane) {
  const int m  = m0 + (lane & 15);
  const int kb = (lane & 16) ? 8 : 0;
  const bf16* p = As + m * LDK + kb;
  union { v16bf v; v8bf h[2]; } u;
  u.h[0] = *(const v8bf*)(p);
  u.h[1] = *(const v8bf*)(p + 16);
  return u.v;
}
__device__ __forceinline__ v16bf frag_b(const bf16* Bs, int n0, int lane) {
  const int n = n0 + (lane & 15);
  const int g = (lane & 16) ? 8 : 0;
  const bf16* p = Bs + n * LDK + g;
  union { v16bf v; v8bf h[2]; } u;
  u.h[0] = *(const v8bf*)(p);
  u.h[1] = *(const v8bf*)(p + 16);
  return u.v;
}

#define WMMA_BF16(a, b, c) \
  __builtin_amdgcn_wmma_f32_16x16x32_bf16(false, (a), false, (b), (short)0, (c), false, false)

// ---------------------------------------------------------------------------
// 1. causal depthwise conv (K=5, left pad 4), fp32 -> bf16
// ---------------------------------------------------------------------------
__global__ __launch_bounds__(256) void k_conv(const float* __restrict__ tgt,
                                              const float* __restrict__ cw,
                                              const float* __restrict__ cb,
                                              bf16* __restrict__ t_bf) {
  const int idx = blockIdx.x * blockDim.x + threadIdx.x; // over B*T*D
  const int d = idx % D_MODEL;
  const int t = (idx / D_MODEL) % SEQ;
  const int b = idx / (D_MODEL * SEQ);
  float acc = cb[d];
#pragma unroll
  for (int k = 0; k < 5; ++k) {
    const int ts = t - 4 + k;
    if (ts >= 0) acc += tgt[((size_t)b * SEQ + ts) * D_MODEL + d] * cw[d * 5 + k];
  }
  t_bf[idx] = (bf16)acc;
}

// ---------------------------------------------------------------------------
// 2. h = silu(x @ Wg^T) * (x @ Wu^T)   M=8192 N=4096 K=1024, fused epilogue
// ---------------------------------------------------------------------------
__global__ __launch_bounds__(256) void k_gemm_swiglu(const float* __restrict__ x,
                                                     const float* __restrict__ wgu,
                                                     bf16* __restrict__ h) {
  __shared__ bf16 As[BM * LDK];
  __shared__ bf16 BsG[BN * LDK];
  __shared__ bf16 BsU[BN * LDK];
  const int tid  = threadIdx.x;
  const int lane = tid & 31;
  const int wave = tid >> 5;
  const int wm = wave & 3, wn = wave >> 2;
  const int m_blk = blockIdx.y * BM;
  const int n_blk = blockIdx.x * BN;
  const int K = D_MODEL;

  v8f accG[2][2] = {};
  v8f accU[2][2] = {};

  for (int k0 = 0; k0 < K; k0 += BK) {
    __syncthreads();
#pragma unroll
    for (int it = 0; it < 4; ++it) {               // A: 128x32 f32 -> bf16
      const int idx = tid + it * NTHREADS;
      const int k4 = (idx & 7) * 4, m = idx >> 3;
      v4f v = *(const v4f*)(x + (size_t)(m_blk + m) * K + k0 + k4);
      bf16* dst = As + m * LDK + k4;
      dst[0] = (bf16)v[0]; dst[1] = (bf16)v[1]; dst[2] = (bf16)v[2]; dst[3] = (bf16)v[3];
    }
#pragma unroll
    for (int it = 0; it < 2; ++it) {               // B gate & up tiles
      const int idx = tid + it * NTHREADS;
      const int k4 = (idx & 7) * 4, n = idx >> 3;
      v4f vg = *(const v4f*)(wgu + (size_t)(n_blk + n) * K + k0 + k4);
      v4f vu = *(const v4f*)(wgu + (size_t)(n_blk + n + FFN) * K + k0 + k4);
      bf16* dg = BsG + n * LDK + k4;
      bf16* du = BsU + n * LDK + k4;
      dg[0] = (bf16)vg[0]; dg[1] = (bf16)vg[1]; dg[2] = (bf16)vg[2]; dg[3] = (bf16)vg[3];
      du[0] = (bf16)vu[0]; du[1] = (bf16)vu[1]; du[2] = (bf16)vu[2]; du[3] = (bf16)vu[3];
    }
    if (k0 + BK < K)                               // global_prefetch_b8 next A tile
      __builtin_prefetch(x + (size_t)(m_blk + (tid >> 3)) * K + k0 + BK + (tid & 7) * 4, 0, 1);
    __syncthreads();

    v16bf a0 = frag_a(As, wm * 32, lane);
    v16bf a1 = frag_a(As, wm * 32 + 16, lane);
    v16bf bg0 = frag_b(BsG, wn * 32, lane);
    v16bf bg1 = frag_b(BsG, wn * 32 + 16, lane);
    v16bf bu0 = frag_b(BsU, wn * 32, lane);
    v16bf bu1 = frag_b(BsU, wn * 32 + 16, lane);
    accG[0][0] = WMMA_BF16(a0, bg0, accG[0][0]);
    accG[0][1] = WMMA_BF16(a0, bg1, accG[0][1]);
    accG[1][0] = WMMA_BF16(a1, bg0, accG[1][0]);
    accG[1][1] = WMMA_BF16(a1, bg1, accG[1][1]);
    accU[0][0] = WMMA_BF16(a0, bu0, accU[0][0]);
    accU[0][1] = WMMA_BF16(a0, bu1, accU[0][1]);
    accU[1][0] = WMMA_BF16(a1, bu0, accU[1][0]);
    accU[1][1] = WMMA_BF16(a1, bu1, accU[1][1]);
  }

  const int ncol = lane & 15;
  const int mrow = (lane >> 4) * 8;
#pragma unroll
  for (int i = 0; i < 2; ++i)
#pragma unroll
    for (int j = 0; j < 2; ++j)
#pragma unroll
      for (int r = 0; r < 8; ++r) {
        const int mg = m_blk + wm * 32 + i * 16 + mrow + r;
        const int ng = n_blk + wn * 32 + j * 16 + ncol;
        const float g = accG[i][j][r], u = accU[i][j][r];
        const float s = g / (1.0f + __expf(-g));
        h[(size_t)mg * FFN + ng] = (bf16)(s * u);
      }
}

// ---------------------------------------------------------------------------
// 3. t_proj = t_chunked[:, :15] @ ttt_proj^T   M=7680 N=1024 K=1024
//    A tile: per-lane GLOBAL_LOAD_ASYNC_TO_LDS_B128 (ASYNCcnt path)
// ---------------------------------------------------------------------------
__global__ __launch_bounds__(256) void k_gemm_tproj(const bf16* __restrict__ t,
                                                    const float* __restrict__ proj,
                                                    bf16* __restrict__ tp) {
  __shared__ bf16 As[BM * LDK];
  __shared__ bf16 Bs[BN * LDK];
  const int tid = threadIdx.x, lane = tid & 31, wave = tid >> 5;
  const int wm = wave & 3, wn = wave >> 2;
  const int r_blk = blockIdx.y * BM;               // row in (B, 15*256) space
  const int b = r_blk / (15 * CHUNK);
  const int a_row0 = b * SEQ + (r_blk - b * 15 * CHUNK);
  const int n_blk = blockIdx.x * BN;
  const int K = D_MODEL;

  v8f acc[2][2] = {};
  for (int k0 = 0; k0 < K; k0 += BK) {
    __syncthreads();                               // all waves done reading LDS
#pragma unroll
    for (int it = 0; it < 2; ++it) {               // A: async copy 16B/lane
      const int idx = tid + it * NTHREADS;
      const int k8 = (idx & 3) * 8, m = idx >> 2;
      async_copy_b128(t + (size_t)(a_row0 + m) * D_MODEL + k0 + k8,
                      lds_off_of(As + m * LDK + k8));
    }
#pragma unroll
    for (int it = 0; it < 2; ++it) {               // B: ttt_proj fp32 -> bf16
      const int idx = tid + it * NTHREADS;
      const int k4 = (idx & 7) * 4, n = idx >> 3;
      v4f v = *(const v4f*)(proj + (size_t)(n_blk + n) * K + k0 + k4);
      bf16* d = Bs + n * LDK + k4;
      d[0] = (bf16)v[0]; d[1] = (bf16)v[1]; d[2] = (bf16)v[2]; d[3] = (bf16)v[3];
    }
    wait_async0();                                 // own async copies landed
    __syncthreads();                               // everyone's tile visible
    v16bf a0 = frag_a(As, wm * 32, lane);
    v16bf a1 = frag_a(As, wm * 32 + 16, lane);
    v16bf b0 = frag_b(Bs, wn * 32, lane);
    v16bf b1 = frag_b(Bs, wn * 32 + 16, lane);
    acc[0][0] = WMMA_BF16(a0, b0, acc[0][0]);
    acc[0][1] = WMMA_BF16(a0, b1, acc[0][1]);
    acc[1][0] = WMMA_BF16(a1, b0, acc[1][0]);
    acc[1][1] = WMMA_BF16(a1, b1, acc[1][1]);
  }
  const int ncol = lane & 15, mrow = (lane >> 4) * 8;
#pragma unroll
  for (int i = 0; i < 2; ++i)
#pragma unroll
    for (int j = 0; j < 2; ++j)
#pragma unroll
      for (int r = 0; r < 8; ++r) {
        const int mg = r_blk + wm * 32 + i * 16 + mrow + r;
        const int ng = n_blk + wn * 32 + j * 16 + ncol;
        tp[(size_t)mg * D_MODEL + ng] = (bf16)acc[i][j][r];
      }
}

// ---------------------------------------------------------------------------
// 4. d_down = t_proj_s^T @ h_s   (per b)  M=1024(D) N=4096(F) K=256(C)
// ---------------------------------------------------------------------------
__global__ __launch_bounds__(256) void k_gemm_ddown(const bf16* __restrict__ tp,
                                                    const bf16* __restrict__ h,
                                                    float* __restrict__ dtmp, int s) {
  __shared__ bf16 As[BM * LDK];
  __shared__ bf16 Bs[BN * LDK];
  const int tid = threadIdx.x, lane = tid & 31, wave = tid >> 5;
  const int wm = wave & 3, wn = wave >> 2;
  const int b = blockIdx.z;
  const int m_blk = blockIdx.y * BM;               // over D
  const int n_blk = blockIdx.x * BN;               // over F
  const int tp_row0 = b * 15 * CHUNK + s * CHUNK;
  const int h_row0  = b * SEQ + s * CHUNK;

  v8f acc[2][2] = {};
  for (int k0 = 0; k0 < CHUNK; k0 += BK) {
    __syncthreads();
#pragma unroll
    for (int it = 0; it < 2; ++it) {               // A = tp^T: gather, scatter to LDS
      const int idx = tid + it * NTHREADS;         // 0..511
      const int m8 = (idx & 15) * 8, k = idx >> 4;
      v8bf v = *(const v8bf*)(tp + (size_t)(tp_row0 + k0 + k) * D_MODEL + m_blk + m8);
#pragma unroll
      for (int j = 0; j < 8; ++j) As[(m8 + j) * LDK + k] = v[j];
    }
    {                                              // B = h: gather, scatter to LDS
      const int idx = tid;                         // 0..255
      const int n8 = (idx & 7) * 8, k = idx >> 3;
      v8bf v = *(const v8bf*)(h + (size_t)(h_row0 + k0 + k) * FFN + n_blk + n8);
#pragma unroll
      for (int j = 0; j < 8; ++j) Bs[(n8 + j) * LDK + k] = v[j];
    }
    __syncthreads();
    v16bf a0 = frag_a(As, wm * 32, lane);
    v16bf a1 = frag_a(As, wm * 32 + 16, lane);
    v16bf b0 = frag_b(Bs, wn * 32, lane);
    v16bf b1 = frag_b(Bs, wn * 32 + 16, lane);
    acc[0][0] = WMMA_BF16(a0, b0, acc[0][0]);
    acc[0][1] = WMMA_BF16(a0, b1, acc[0][1]);
    acc[1][0] = WMMA_BF16(a1, b0, acc[1][0]);
    acc[1][1] = WMMA_BF16(a1, b1, acc[1][1]);
  }
  const int ncol = lane & 15, mrow = (lane >> 4) * 8;
#pragma unroll
  for (int i = 0; i < 2; ++i)
#pragma unroll
    for (int j = 0; j < 2; ++j)
#pragma unroll
      for (int r = 0; r < 8; ++r) {
        const int mg = m_blk + wm * 32 + i * 16 + mrow + r;
        const int ng = n_blk + wn * 32 + j * 16 + ncol;
        dtmp[((size_t)b * D_MODEL + mg) * FFN + ng] = acc[i][j][r];
      }
}

// ---------------------------------------------------------------------------
// 5. out_chunk = h_chunk @ W[b]^T   M=512(B*C) N=1024(D) K=4096(F)
//    A tile: one TDM tensor_load_to_lds per WG per K-step (TENSORcnt path).
//    D# descriptor: 2D tile 128 rows x 32 bf16, row stride FFN, LDS padding
//    16B after every 64B row chunk to reproduce the LDK=40 LDS layout.
// ---------------------------------------------------------------------------
__global__ __launch_bounds__(256) void k_gemm_out(const bf16* __restrict__ h,
                                                  const float* __restrict__ W,
                                                  float* __restrict__ out, int s) {
  __shared__ bf16 As[BM * LDK];
  __shared__ bf16 Bs[BN * LDK];
  const int tid = threadIdx.x, lane = tid & 31, wave = tid >> 5;
  const int wm = wave & 3, wn = wave >> 2;
  const int m_blk = blockIdx.y * BM;               // over B*CHUNK
  const int b = m_blk >> 8;                        // BM=128 never straddles b
  const int c0 = m_blk & 255;
  const int n_blk = blockIdx.x * BN;               // over D
  const int h_row0 = b * SEQ + s * CHUNK + c0;
  const int K = FFN;

  v8f acc[2][2] = {};
  for (int k0 = 0; k0 < K; k0 += BK) {
    __syncthreads();                               // all waves done reading LDS
#if __has_builtin(__builtin_amdgcn_tensor_load_to_lds)
    if (tid < 32) {                                // wave 0 issues the DMA
      const unsigned long long ga =
          (unsigned long long)(uintptr_t)(h + (size_t)h_row0 * FFN + k0);
      v4u g0;
      g0[0] = 1u;                                  // count=1, no gather
      g0[1] = lds_off_of(As);                      // lds_addr
      g0[2] = (unsigned)(ga & 0xFFFFFFFFu);        // global_addr[31:0]
      g0[3] = (unsigned)((ga >> 32) & 0x01FFFFFFu) // global_addr[56:32]
              | (2u << 30);                        // type = 2 (image)
      v8i_ g1;
      g1[0] = (int)((1u << 16)                     // data_size = 2B
                  | (1u << 20)                     // pad_enable
                  | (3u << 22)                     // pad_interval: 16 DWORDs (64B)
                  | (3u << 25));                   // pad_amount: 4 DWORDs (16B)
      g1[1] = (int)((unsigned)FFN << 16);          // tensor_dim0[15:0] = 4096
      g1[2] = (int)(0x8000u << 16);                // tensor_dim1[15:0] = 32768
      g1[3] = (int)(32u << 16);                    // tile_dim0 = 32 elems
      g1[4] = 128;                                 // tile_dim1 = 128 rows
      g1[5] = FFN;                                 // tensor_dim0_stride = 4096
      g1[6] = 0;
      g1[7] = 0;
      v4i_ gz4 = {0, 0, 0, 0};
      v8i_ gz8 = {0, 0, 0, 0, 0, 0, 0, 0};
      __builtin_amdgcn_tensor_load_to_lds(g0, g1, gz4, gz4, gz8, 0);
    }
#else
#pragma unroll
    for (int it = 0; it < 2; ++it) {               // fallback: per-lane async copy
      const int idx = tid + it * NTHREADS;
      const int k8 = (idx & 3) * 8, m = idx >> 2;
      async_copy_b128(h + (size_t)(h_row0 + m) * FFN + k0 + k8,
                      lds_off_of(As + m * LDK + k8));
    }
#endif
#pragma unroll
    for (int it = 0; it < 2; ++it) {               // B: running W fp32 -> bf16
      const int idx = tid + it * NTHREADS;
      const int k4 = (idx & 7) * 4, n = idx >> 3;
      v4f v = *(const v4f*)(W + ((size_t)b * D_MODEL + n_blk + n) * FFN + k0 + k4);
      bf16* d = Bs + n * LDK + k4;
      d[0] = (bf16)v[0]; d[1] = (bf16)v[1]; d[2] = (bf16)v[2]; d[3] = (bf16)v[3];
    }
    if (k0 + BK < K)
      __builtin_prefetch(W + ((size_t)b * D_MODEL + n_blk + (tid >> 3)) * FFN + k0 + BK + (tid & 7) * 4, 0, 1);
#if __has_builtin(__builtin_amdgcn_tensor_load_to_lds)
    __builtin_amdgcn_s_wait_tensorcnt(0);          // wave 0's DMA complete
#else
    wait_async0();
#endif
    __syncthreads();                               // tile visible to all waves
    v16bf a0 = frag_a(As, wm * 32, lane);
    v16bf a1 = frag_a(As, wm * 32 + 16, lane);
    v16bf b0 = frag_b(Bs, wn * 32, lane);
    v16bf b1 = frag_b(Bs, wn * 32 + 16, lane);
    acc[0][0] = WMMA_BF16(a0, b0, acc[0][0]);
    acc[0][1] = WMMA_BF16(a0, b1, acc[0][1]);
    acc[1][0] = WMMA_BF16(a1, b0, acc[1][0]);
    acc[1][1] = WMMA_BF16(a1, b1, acc[1][1]);
  }
  const int ncol = lane & 15, mrow = (lane >> 4) * 8;
#pragma unroll
  for (int i = 0; i < 2; ++i)
#pragma unroll
    for (int j = 0; j < 2; ++j)
#pragma unroll
      for (int r = 0; r < 8; ++r) {
        const int c = c0 + wm * 32 + i * 16 + mrow + r;
        const int ng = n_blk + wn * 32 + j * 16 + ncol;
        out[((size_t)b * SEQ + s * CHUNK + c) * D_MODEL + ng] = acc[i][j][r];
      }
}

// ---------------------------------------------------------------------------
// helpers: W init / W += lr * d_tmp
// ---------------------------------------------------------------------------
__global__ __launch_bounds__(256) void k_initW(const float* __restrict__ wd,
                                               float* __restrict__ W) {
  const int i = blockIdx.x * blockDim.x + threadIdx.x;   // over D*F
  const float v = wd[i];
  W[i] = v;
  W[(size_t)D_MODEL * FFN + i] = v;
}

__global__ __launch_bounds__(256) void k_axpy(float* __restrict__ W,
                                              const float* __restrict__ d) {
  const int i = blockIdx.x * blockDim.x + threadIdx.x;   // over B*D*F
  W[i] += TTT_LR * d[i];
}

// ---------------------------------------------------------------------------
extern "C" void kernel_launch(void* const* d_in, const int* in_sizes, int n_in,
                              void* d_out, int out_size, void* d_ws, size_t ws_size,
                              hipStream_t stream) {
  const float* x   = (const float*)d_in[0];
  const float* tgt = (const float*)d_in[1];
  const float* wgu = (const float*)d_in[2];
  const float* wdn = (const float*)d_in[3];
  const float* prj = (const float*)d_in[4];
  const float* cw  = (const float*)d_in[5];
  const float* cb  = (const float*)d_in[6];
  float* out = (float*)d_out;

  // workspace layout (bytes):
  char* ws = (char*)d_ws;
  size_t off = 0;
  bf16* h  = (bf16*)(ws + off); off += (size_t)BATCH * SEQ * FFN * 2;           // 64 MB
  bf16* t  = (bf16*)(ws + off); off += (size_t)BATCH * SEQ * D_MODEL * 2;       // 16 MB
  bf16* tp = (bf16*)(ws + off); off += (size_t)BATCH * 15 * CHUNK * D_MODEL * 2;// 15 MB
  float* W  = (float*)(ws + off); off += (size_t)BATCH * D_MODEL * FFN * 4;     // 32 MB
  float* dt = (float*)(ws + off); off += (size_t)BATCH * D_MODEL * FFN * 4;     // 32 MB
  (void)ws_size; (void)in_sizes; (void)n_in; (void)out_size;

  // 1. conv (B*T*D threads)
  k_conv<<<(BATCH * SEQ * D_MODEL) / 256, 256, 0, stream>>>(tgt, cw, cb, t);
  // 2. SwiGLU GEMM: grid (N/64, M/128) = (64, 64)
  k_gemm_swiglu<<<dim3(FFN / BN, (BATCH * SEQ) / BM), 256, 0, stream>>>(x, wgu, h);
  // 3. t_proj GEMM: grid (1024/64, 7680/128) = (16, 60)
  k_gemm_tproj<<<dim3(D_MODEL / BN, (BATCH * 15 * CHUNK) / BM), 256, 0, stream>>>(t, prj, tp);
  // 4. init running W
  k_initW<<<(D_MODEL * FFN) / 256, 256, 0, stream>>>(wdn, W);
  // 5. sequential cumsum over chunks (stream order carries the dependency)
  for (int s = 0; s < NC; ++s) {
    k_gemm_out<<<dim3(D_MODEL / BN, (BATCH * CHUNK) / BM), 256, 0, stream>>>(h, W, out, s);
    if (s < NC - 1) {
      k_gemm_ddown<<<dim3(FFN / BN, D_MODEL / BM, BATCH), 256, 0, stream>>>(tp, h, dt, s);
      k_axpy<<<(BATCH * D_MODEL * FFN) / 256, 256, 0, stream>>>(W, dt);
    }
  }
}
